// GDNSequenceModel_21827023798781
// MI455X (gfx1250) — matrix operand order repaired and checked by reading, not compiled, not run on hardware
//
#include <hip/hip_runtime.h>
#include <hip/hip_bf16.h>
#include <math.h>

// ---------------------------------------------------------------------------
// Problem constants (from reference): B=4 T=4096 D=1024 H=8 K=128 V=128 F=4096
// ---------------------------------------------------------------------------
static constexpr int Bn = 4, Tn = 4096, Dn = 1024, Hn = 8, Kn = 128, Vv = 128, Fn = 4096;
static constexpr int Mn = Bn * Tn; // 16384 tokens

// ---------------------------------------------------------------------------
// CDNA5 WMMA plumbing (bf16 16x16x32 -> f32 accum), wave32
// ---------------------------------------------------------------------------
typedef __attribute__((ext_vector_type(16))) __bf16 v16bf;
typedef __attribute__((ext_vector_type(8)))  float  v8f;

union FragB { unsigned u[8]; v16bf v; };

__device__ __forceinline__ v8f wmma_bf16(const FragB& a, const FragB& b, v8f c) {
  return __builtin_amdgcn_wmma_f32_16x16x32_bf16(false, a.v, false, b.v, (short)0, c, false, false);
}

// A-operand (16x32 bf16): lane half h, VGPR g -> K-pair index
__device__ __forceinline__ int jpmap(int g, int half) {
  return 4 * half + g + ((g & 4) ? 4 : 0);
}

__device__ __forceinline__ unsigned short f2bf(float f) {
  unsigned u = __float_as_uint(f);
  u += 0x7fffu + ((u >> 16) & 1u); // RNE
  return (unsigned short)(u >> 16);
}
__device__ __forceinline__ float bf2f(unsigned short h) {
  return __uint_as_float(((unsigned)h) << 16);
}
__device__ __forceinline__ unsigned pack2(unsigned short lo, unsigned short hi) {
  return (unsigned)lo | ((unsigned)hi << 16);
}
__device__ __forceinline__ float siluf(float x) { return x / (1.f + expf(-x)); }
__device__ __forceinline__ float geluf(float x) { return 0.5f * x * (1.f + erff(x * 0.70710678118654752f)); }

// ---------------------------------------------------------------------------
// fp32 -> bf16 weight conversion
// ---------------------------------------------------------------------------
__global__ void cvt_f32_bf16(const float* __restrict__ src, unsigned short* __restrict__ dst, int n) {
  int i = blockIdx.x * blockDim.x + threadIdx.x;
  if (i < n) dst[i] = f2bf(src[i]);
}

// ---------------------------------------------------------------------------
// LayerNorm (one 256-thread block per token row of D=1024), bf16 out
// ---------------------------------------------------------------------------
__global__ __launch_bounds__(256) void ln_kernel(const float* __restrict__ x,
                                                 const float* __restrict__ w,
                                                 const float* __restrict__ b,
                                                 unsigned short* __restrict__ out) {
  __shared__ float s1[256], s2[256];
  int row = blockIdx.x, tid = threadIdx.x;
  const float* xr = x + (size_t)row * Dn;
  float s = 0.f, ss = 0.f;
  for (int i = tid; i < Dn; i += 256) { float v = xr[i]; s += v; ss += v * v; }
  s1[tid] = s; s2[tid] = ss; __syncthreads();
  for (int st = 128; st > 0; st >>= 1) {
    if (tid < st) { s1[tid] += s1[tid + st]; s2[tid] += s2[tid + st]; }
    __syncthreads();
  }
  float m = s1[0] * (1.f / Dn);
  float var = s2[0] * (1.f / Dn) - m * m;
  float inv = rsqrtf(var + 1e-5f);
  for (int i = tid; i < Dn; i += 256)
    out[(size_t)row * Dn + i] = f2bf((xr[i] - m) * inv * w[i] + b[i]);
}

// ---------------------------------------------------------------------------
// Gate projections: beta = sigmoid(n@Wb), alpha = sigmoid(n@Wa)  (N=8 each)
// one block per token
// ---------------------------------------------------------------------------
__global__ __launch_bounds__(256) void gates_kernel(const unsigned short* __restrict__ xn,
                                                    const float* __restrict__ Wb,
                                                    const float* __restrict__ Wa,
                                                    float* __restrict__ beta,
                                                    float* __restrict__ alpha) {
  int row = blockIdx.x, tid = threadIdx.x;
  int w = tid >> 5, l = tid & 31;
  const unsigned short* xr = xn + (size_t)row * Dn;
  float acc[16];
  for (int o = 0; o < 16; ++o) acc[o] = 0.f;
  for (int d = tid; d < Dn; d += 256) {
    float xv = bf2f(xr[d]);
    for (int o = 0; o < 8; ++o) {
      acc[o]     += xv * Wb[d * Hn + o];
      acc[8 + o] += xv * Wa[d * Hn + o];
    }
  }
  for (int o = 0; o < 16; ++o)
    for (int m = 16; m; m >>= 1) acc[o] += __shfl_xor(acc[o], m, 32);
  __shared__ float part[8][16];
  if (l == 0) for (int o = 0; o < 16; ++o) part[w][o] = acc[o];
  __syncthreads();
  if (tid < 16) {
    float s = 0.f;
    for (int ww = 0; ww < 8; ++ww) s += part[ww][tid];
    float sig = 1.f / (1.f + expf(-s));
    if (tid < 8) beta[(size_t)row * Hn + tid] = sig;
    else         alpha[(size_t)row * Hn + (tid - 8)] = sig;
  }
}

// ---------------------------------------------------------------------------
// Per-head L2 norm (in-place on bf16, silu already applied): one wave per head
// ---------------------------------------------------------------------------
__global__ __launch_bounds__(256) void l2norm_kernel(unsigned short* __restrict__ q) {
  int row = blockIdx.x;
  int w = threadIdx.x >> 5, l = threadIdx.x & 31;
  unsigned short* p = q + ((size_t)row * Hn + w) * Kn;
  float v[4]; float ss = 0.f;
  for (int i = 0; i < 4; ++i) { v[i] = bf2f(p[l + 32 * i]); ss += v[i] * v[i]; }
  for (int m = 16; m; m >>= 1) ss += __shfl_xor(ss, m, 32);
  float inv = rsqrtf(ss + 1e-6f);
  for (int i = 0; i < 4; ++i) p[l + 32 * i] = f2bf(v[i] * inv);
}

// ---------------------------------------------------------------------------
// bf16 WMMA GEMM: C[M,N] = epi(A[M,Kd] @ W[Kd,N]); 128x128 block, 8 waves,
// K-step 32. LDS holds operands pre-packed as bf16 pairs (uint).
// ---------------------------------------------------------------------------
enum { EPI_SILU = 0, EPI_GELU_BIAS = 1, EPI_RES = 2, EPI_ACC_BIAS = 3 };

template <int EPI>
__global__ __launch_bounds__(256, 1) void gemm_bf16_kernel(
    const unsigned* __restrict__ A, const unsigned* __restrict__ W,
    int M, int N, int Kd,
    unsigned short* __restrict__ outb, float* __restrict__ outf,
    const float* __restrict__ bias, const float* __restrict__ res) {
  (void)M;
  __shared__ unsigned As[128 * 16]; // [row][K-pair]   (A tile 128x32)
  __shared__ unsigned Bs[32 * 64];  // [Krow][N-pair]  (W tile 32x128)
  int tid = threadIdx.x;
  int w = tid >> 5, l = tid & 31, half = l >> 4, ln = l & 15;
  int m0 = blockIdx.x * 128, n0 = blockIdx.y * 128;
  int Kp = Kd >> 1, Np = N >> 1;
  v8f z = {0.f, 0.f, 0.f, 0.f, 0.f, 0.f, 0.f, 0.f};
  v8f acc[8];
  for (int i = 0; i < 8; ++i) acc[i] = z;
  for (int kk = 0; kk < Kd; kk += 32) {
    for (int r = 0; r < 8; ++r) {
      int idx = tid + 256 * r;
      int ra = idx >> 4, ja = idx & 15;
      As[idx] = A[(size_t)(m0 + ra) * Kp + (kk >> 1) + ja];
      int rb = idx >> 6, jb = idx & 63;
      Bs[idx] = W[(size_t)(kk + rb) * Np + (n0 >> 1) + jb];
    }
    if (kk + 32 < Kd)
      __builtin_prefetch(&A[(size_t)(m0 + (tid >> 4)) * Kp + ((kk + 32) >> 1)], 0, 0);
    __syncthreads();
    FragB a;
    for (int g = 0; g < 8; ++g) a.u[g] = As[(16 * w + ln) * 16 + jpmap(g, half)];
    for (int nt = 0; nt < 8; ++nt) {
      FragB bb;
      for (int g = 0; g < 8; ++g) bb.u[g] = pack2((unsigned short)(Bs[(2 * jpmap(g, half)) * 64 + nt * 8 + (ln >> 1)] >> ((ln & 1) * 16)), (unsigned short)(Bs[(2 * jpmap(g, half) + 1) * 64 + nt * 8 + (ln >> 1)] >> ((ln & 1) * 16)));
      acc[nt] = wmma_bf16(a, bb, acc[nt]);
    }
    __syncthreads();
  }
  for (int nt = 0; nt < 8; ++nt) {
    for (int r = 0; r < 8; ++r) {
      int m = m0 + 16 * w + r + 8 * half;
      int n = n0 + 16 * nt + ln;
      size_t idx = (size_t)m * N + n;
      float v = acc[nt][r];
      if (EPI == EPI_SILU)           { outb[idx] = f2bf(siluf(v)); }
      else if (EPI == EPI_GELU_BIAS) { outb[idx] = f2bf(geluf(v + bias[n])); }
      else if (EPI == EPI_RES)       { outf[idx] = res[idx] + v; }
      else                           { outf[idx] = outf[idx] + v + bias[n]; }
    }
  }
}

// ---------------------------------------------------------------------------
// Exact chunked delta rule (chunk C=16), one workgroup (8 waves) per (b,h).
// State S[K=128][V=128] fp32 in LDS; every chunk-level matmul is a bf16 WMMA.
// ---------------------------------------------------------------------------
static constexpr unsigned RC_SMEM = 150720;

__global__ __launch_bounds__(256, 1) void gdn_recurrence_kernel(
    const unsigned* __restrict__ qg, const unsigned* __restrict__ kg,
    const unsigned* __restrict__ vg, const float* __restrict__ beta,
    const float* __restrict__ alpha, unsigned short* __restrict__ og) {
  extern __shared__ unsigned char smem[];
  float*    S   = (float*)(smem);               // 128x128 f32            65536 B
  unsigned* Sb  = (unsigned*)(smem + 65536);    // 128x64 bf16-pairs      32768 B
  unsigned* Qc  = (unsigned*)(smem + 98304);    // 16x64 bf16-pairs        4096 B
  unsigned* Kc  = (unsigned*)(smem + 102400);   // 16x64                   4096 B
  unsigned* Kt  = (unsigned*)(smem + 106496);   // 128x16 (K^T, pad 32)    8192 B
  float*    Vc  = (float*)(smem + 114688);      // 16x128 f32              8192 B
  float*    Wm  = (float*)(smem + 122880);      // K_c @ S0                8192 B
  float*    Wqm = (float*)(smem + 131072);      // Q_c @ S0                8192 B
  float*    U   = (float*)(smem + 139264);      // 16x128 f32              8192 B
  float*    KK  = (float*)(smem + 147456);      // 16x16                   1024 B
  float*    QK  = (float*)(smem + 148480);      // 16x16                   1024 B
  float*    Ml  = (float*)(smem + 149504);      // 16x16                   1024 B
  float*    gv  = (float*)(smem + 150528);      // 16
  float*    bv  = (float*)(smem + 150592);      // 16
  float*    dv  = (float*)(smem + 150656);      // 16
  // phase-disjoint aliases
  unsigned* Ub = Sb;            // 32x64 bf16-pairs (rows 16..31 zero)
  unsigned* Ud = Sb + 2048;     // 32x64, scaled by d_j = G/g_j
  unsigned* Ab = Qc;            // 16x16 bf16-pairs (masked A, pad 32)

  int tid = threadIdx.x;
  int w = tid >> 5, l = tid & 31, half = l >> 4, ln = l & 15;
  int bb = blockIdx.x >> 3, hh = blockIdx.x & 7;
  size_t rbase = ((size_t)bb * Tn * Hn + hh) * (Kn / 2); // u32 row base; +512 per token
  v8f z = {0.f, 0.f, 0.f, 0.f, 0.f, 0.f, 0.f, 0.f};

  for (int i = tid; i < 128 * 128; i += 256) S[i] = 0.f;
  __syncthreads();

  for (int c = 0; c < Tn / 16; ++c) {
    int t0 = c * 16;
    // ---- stage chunk q/k/v + gates -------------------------------------
    for (int i = tid; i < 1024; i += 256) {
      int j = i >> 6, p = i & 63;
      size_t ro = rbase + (size_t)(t0 + j) * (Hn * Kn / 2);
      Qc[i] = qg[ro + p];
      Kc[i] = kg[ro + p];
      unsigned uv = vg[ro + p];
      Vc[j * 128 + 2 * p]     = bf2f((unsigned short)(uv & 0xffffu));
      Vc[j * 128 + 2 * p + 1] = bf2f((unsigned short)(uv >> 16));
    }
    if (tid < 16) {
      size_t gi = ((size_t)bb * Tn + t0 + tid) * Hn + hh;
      bv[tid] = beta[gi];
      gv[tid] = alpha[gi]; // raw a_j; cumprod below
    }
    if (c + 1 < Tn / 16)
      __builtin_prefetch(&qg[rbase + (size_t)(t0 + 16) * (Hn * Kn / 2)], 0, 0);
    __syncthreads();
    // ---- K^T (pad chunk dim to 32) + bf16 stage of S + cumulative decay
    for (int i = tid; i < 2048; i += 256) {
      int kk = i >> 4, jp = i & 15;
      unsigned val = 0;
      if (jp < 8) {
        unsigned u0 = Kc[(2 * jp) * 64 + (kk >> 1)];
        unsigned u1 = Kc[(2 * jp + 1) * 64 + (kk >> 1)];
        unsigned short lo = (kk & 1) ? (unsigned short)(u0 >> 16) : (unsigned short)(u0 & 0xffffu);
        unsigned short hi = (kk & 1) ? (unsigned short)(u1 >> 16) : (unsigned short)(u1 & 0xffffu);
        val = pack2(lo, hi);
      }
      Kt[i] = val;
    }
    for (int i = tid; i < 8192; i += 256) {
      int kk = i >> 6, p = i & 63;
      Sb[i] = pack2(f2bf(S[kk * 128 + 2 * p]), f2bf(S[kk * 128 + 2 * p + 1]));
    }
    if (tid == 0) {
      float g = 1.f;
      for (int j = 0; j < 16; ++j) { g *= gv[j]; gv[j] = g; }
      for (int j = 0; j < 16; ++j) dv[j] = gv[15] / gv[j];
    }
    __syncthreads();
    // ---- phase 1: W = K_c@S0, Wq = Q_c@S0, KK = K_c K_c^T, QK = Q_c K_c^T
    {
      v8f accW = z, accWq = z, accKK = z, accQK = z;
      for (int s = 0; s < 4; ++s) {
        FragB aK, aQ, bS;
        for (int g = 0; g < 8; ++g) {
          int jp = jpmap(g, half);
          aK.u[g] = Kc[ln * 64 + 16 * s + jp];
          aQ.u[g] = Qc[ln * 64 + 16 * s + jp];
          bS.u[g] = pack2((unsigned short)(Sb[(32 * s + 2 * jp) * 64 + 8 * w + (ln >> 1)] >> ((ln & 1) * 16)), (unsigned short)(Sb[(32 * s + 2 * jp + 1) * 64 + 8 * w + (ln >> 1)] >> ((ln & 1) * 16)));
        }
        accW  = wmma_bf16(aK, bS, accW);
        accWq = wmma_bf16(aQ, bS, accWq);
        if (w == 0) {
          FragB bK;
          for (int g = 0; g < 8; ++g) bK.u[g] = aK.u[g];
          accKK = wmma_bf16(aK, bK, accKK);
        }
        if (w == 1) {
          FragB bK;
          for (int g = 0; g < 8; ++g) bK.u[g] = aK.u[g];
          accQK = wmma_bf16(aQ, bK, accQK);
        }
      }
      for (int r = 0; r < 8; ++r) {
        int j = r + 8 * half;
        Wm[j * 128 + 16 * w + ln]  = accW[r];
        Wqm[j * 128 + 16 * w + ln] = accWq[r];
        if (w == 0) KK[j * 16 + ln] = accKK[r];
        if (w == 1) QK[j * 16 + ln] = accQK[r];
      }
    }
    __syncthreads();
    // ---- phase 2a: masked M, masked+packed A, rhs of U ------------------
    {
      int j = tid >> 4, i = tid & 15;
      Ml[tid] = (i < j) ? bv[j] * (gv[j] / gv[i]) * KK[j * 16 + i] : 0.f;
      unsigned val = 0;
      if (i < 8) {
        int i0 = 2 * i, i1 = 2 * i + 1;
        float A0 = (i0 <= j) ? (gv[j] / gv[i0]) * QK[j * 16 + i0] : 0.f;
        float A1 = (i1 <= j) ? (gv[j] / gv[i1]) * QK[j * 16 + i1] : 0.f;
        val = pack2(f2bf(A0), f2bf(A1));
      }
      Ab[tid] = val;
    }
    for (int i = tid; i < 2048; i += 256) {
      int j = i >> 7;
      U[i] = bv[j] * Vc[i] - bv[j] * gv[j] * Wm[i];
    }
    __syncthreads();
    // ---- phase 2b: forward substitution (columns independent) -----------
    if (tid < 128) {
      int v = tid;
      for (int j = 1; j < 16; ++j) {
        float s = 0.f;
        for (int i = 0; i < j; ++i) s += Ml[j * 16 + i] * U[i * 128 + v];
        U[j * 128 + v] -= s;
      }
    }
    __syncthreads();
    // ---- phase 2c: pack U and diag(d)U to bf16 (pad rows to 32) ---------
    for (int i = tid; i < 2048; i += 256) {
      int j = i >> 6, p = i & 63;
      unsigned ub = 0, ud = 0;
      if (j < 16) {
        float u0 = U[j * 128 + 2 * p], u1 = U[j * 128 + 2 * p + 1];
        ub = pack2(f2bf(u0), f2bf(u1));
        ud = pack2(f2bf(dv[j] * u0), f2bf(dv[j] * u1));
      }
      Ub[i] = ub; Ud[i] = ud;
    }
    __syncthreads();
    // ---- phase 3: O = A_mask @ U + diag(g) Wq -> global -----------------
    {
      FragB aA, bU;
      for (int g = 0; g < 8; ++g) {
        aA.u[g] = Ab[ln * 16 + jpmap(g, half)];
        bU.u[g] = pack2((unsigned short)(Ub[(2 * jpmap(g, half)) * 64 + 8 * w + (ln >> 1)] >> ((ln & 1) * 16)), (unsigned short)(Ub[(2 * jpmap(g, half) + 1) * 64 + 8 * w + (ln >> 1)] >> ((ln & 1) * 16)));
      }
      v8f co;
      for (int r = 0; r < 8; ++r) {
        int j = r + 8 * half;
        co[r] = gv[j] * Wqm[j * 128 + 16 * w + ln];
      }
      co = wmma_bf16(aA, bU, co);
      for (int r = 0; r < 8; ++r) {
        int j = r + 8 * half;
        size_t oi = (((size_t)bb * Tn + t0 + j) * Hn + hh) * Vv + 16 * w + ln;
        og[oi] = f2bf(co[r]);
      }
    }
    // ---- phase 4: S = G*S + K^T diag(d) U -------------------------------
    {
      float G = gv[15];
      for (int t = 0; t < 8; ++t) {
        FragB aT, bD;
        for (int g = 0; g < 8; ++g) {
          aT.u[g] = Kt[(16 * t + ln) * 16 + jpmap(g, half)];
          bD.u[g] = pack2((unsigned short)(Ud[(2 * jpmap(g, half)) * 64 + 8 * w + (ln >> 1)] >> ((ln & 1) * 16)), (unsigned short)(Ud[(2 * jpmap(g, half) + 1) * 64 + 8 * w + (ln >> 1)] >> ((ln & 1) * 16)));
        }
        v8f dacc = wmma_bf16(aT, bD, z);
        for (int r = 0; r < 8; ++r) {
          int kk = 16 * t + r + 8 * half;
          int col = 16 * w + ln;
          S[kk * 128 + col] = G * S[kk * 128 + col] + dacc[r];
        }
      }
    }
    __syncthreads();
  }
}

// ---------------------------------------------------------------------------
// Host orchestration
// ---------------------------------------------------------------------------
extern "C" void kernel_launch(void* const* d_in, const int* in_sizes, int n_in,
                              void* d_out, int out_size, void* d_ws, size_t ws_size,
                              hipStream_t stream) {
  (void)in_sizes; (void)n_in; (void)out_size; (void)ws_size;
  const float* x    = (const float*)d_in[0];
  const float* Wq   = (const float*)d_in[1];
  const float* Wk   = (const float*)d_in[2];
  const float* Wv   = (const float*)d_in[3];
  const float* Wb   = (const float*)d_in[4];
  const float* Wa   = (const float*)d_in[5];
  const float* Wo   = (const float*)d_in[6];
  const float* ln1w = (const float*)d_in[7];
  const float* ln1b = (const float*)d_in[8];
  const float* ln2w = (const float*)d_in[9];
  const float* ln2b = (const float*)d_in[10];
  const float* W1   = (const float*)d_in[11];
  const float* b1   = (const float*)d_in[12];
  const float* W2   = (const float*)d_in[13];
  const float* b2   = (const float*)d_in[14];
  float* outF = (float*)d_out;
  char* ws = (char*)d_ws;

  size_t off = 0;
  auto alloc = [&](size_t b) { size_t o = off; off += (b + 255) & ~(size_t)255; return o; };
  unsigned short* wq_b = (unsigned short*)(ws + alloc((size_t)Dn * Hn * Kn * 2));
  unsigned short* wk_b = (unsigned short*)(ws + alloc((size_t)Dn * Hn * Kn * 2));
  unsigned short* wv_b = (unsigned short*)(ws + alloc((size_t)Dn * Hn * Vv * 2));
  unsigned short* wo_b = (unsigned short*)(ws + alloc((size_t)Hn * Vv * Dn * 2));
  unsigned short* w1_b = (unsigned short*)(ws + alloc((size_t)Dn * Fn * 2));
  unsigned short* w2_b = (unsigned short*)(ws + alloc((size_t)Fn * Dn * 2));
  float* beta  = (float*)(ws + alloc((size_t)Mn * Hn * 4));
  float* alpha = (float*)(ws + alloc((size_t)Mn * Hn * 4));
  unsigned short* xn_b = (unsigned short*)(ws + alloc((size_t)Mn * Dn * 2));
  unsigned short* q_b  = (unsigned short*)(ws + alloc((size_t)Mn * Hn * Kn * 2));
  unsigned short* k_b  = (unsigned short*)(ws + alloc((size_t)Mn * Hn * Kn * 2));
  unsigned short* v_b  = (unsigned short*)(ws + alloc((size_t)Mn * Hn * Vv * 2));
  unsigned short* o_b  = (unsigned short*)(ws + alloc((size_t)Mn * Hn * Vv * 2));
  // phase-disjoint overlays: ffn1 (M x F bf16 = 128MB) spans xn+q+k+v; h spans o
  unsigned short* f1_b = xn_b;
  unsigned short* h_b  = o_b;

  auto cvt = [&](const float* s, unsigned short* d, int n) {
    cvt_f32_bf16<<<dim3((n + 255) / 256), dim3(256), 0, stream>>>(s, d, n);
  };
  cvt(Wq, wq_b, Dn * Hn * Kn);
  cvt(Wk, wk_b, Dn * Hn * Kn);
  cvt(Wv, wv_b, Dn * Hn * Vv);
  cvt(Wo, wo_b, Hn * Vv * Dn);
  cvt(W1, w1_b, Dn * Fn);
  cvt(W2, w2_b, Fn * Dn);

  // LN1
  ln_kernel<<<dim3(Mn), dim3(256), 0, stream>>>(x, ln1w, ln1b, xn_b);

  // q/k/v projections with fused SiLU -> bf16
  gemm_bf16_kernel<EPI_SILU><<<dim3(Mn / 128, (Hn * Kn) / 128), dim3(256), 0, stream>>>(
      (const unsigned*)xn_b, (const unsigned*)wq_b, Mn, Hn * Kn, Dn, q_b, nullptr, nullptr, nullptr);
  gemm_bf16_kernel<EPI_SILU><<<dim3(Mn / 128, (Hn * Kn) / 128), dim3(256), 0, stream>>>(
      (const unsigned*)xn_b, (const unsigned*)wk_b, Mn, Hn * Kn, Dn, k_b, nullptr, nullptr, nullptr);
  gemm_bf16_kernel<EPI_SILU><<<dim3(Mn / 128, (Hn * Vv) / 128), dim3(256), 0, stream>>>(
      (const unsigned*)xn_b, (const unsigned*)wv_b, Mn, Hn * Vv, Dn, v_b, nullptr, nullptr, nullptr);

  // gates
  gates_kernel<<<dim3(Mn), dim3(256), 0, stream>>>(xn_b, Wb, Wa, beta, alpha);

  // per-head l2norm on q,k
  l2norm_kernel<<<dim3(Mn), dim3(256), 0, stream>>>(q_b);
  l2norm_kernel<<<dim3(Mn), dim3(256), 0, stream>>>(k_b);

  // chunked delta-rule recurrence (one WG per (b,h))
  gdn_recurrence_kernel<<<dim3(Bn * Hn), dim3(256), RC_SMEM, stream>>>(
      (const unsigned*)q_b, (const unsigned*)k_b, (const unsigned*)v_b, beta, alpha, o_b);

  // x2 = x + o @ Wo  -> d_out (f32)
  gemm_bf16_kernel<EPI_RES><<<dim3(Mn / 128, Dn / 128), dim3(256), 0, stream>>>(
      (const unsigned*)o_b, (const unsigned*)wo_b, Mn, Dn, Hn * Vv, nullptr, outF, nullptr, x);

  // LN2 on x2
  ln_kernel<<<dim3(Mn), dim3(256), 0, stream>>>(outF, ln2w, ln2b, h_b);

  // ffn1 = gelu(h @ W1 + b1) -> bf16
  gemm_bf16_kernel<EPI_GELU_BIAS><<<dim3(Mn / 128, Fn / 128), dim3(256), 0, stream>>>(
      (const unsigned*)h_b, (const unsigned*)w1_b, Mn, Fn, Dn, f1_b, nullptr, b1, nullptr);

  // d_out = x2 + ffn1 @ W2 + b2
  gemm_bf16_kernel<EPI_ACC_BIAS><<<dim3(Mn / 128, Dn / 128), dim3(256), 0, stream>>>(
      (const unsigned*)f1_b, (const unsigned*)w2_b, Mn, Dn, Fn, nullptr, outF, b2, nullptr);
}
